// AEFS_71777493450774
// MI455X (gfx1250) — matrix operand order repaired and checked by reading, not compiled, not run on hardware
//
#include <hip/hip_runtime.h>

#define Bsz 16384
#define Ff  39
#define Vv  30000
#define Dd  16
#define Kk  20
#define FD  (Ff*Dd)     // 624
#define KP0 640         // 624 padded to multiple of 32
#define N0  1024
#define N1  512
#define NT2 (N1/64)     // 8 N-tile groups in GEMM2

typedef __attribute__((ext_vector_type(16))) __bf16 v16bf;
typedef __attribute__((ext_vector_type(8)))  __bf16 v8bf;
typedef __attribute__((ext_vector_type(8)))  float  v8f;

// Two contiguous 16B chunks -> one 16-element bf16 WMMA operand
__device__ __forceinline__ v16bf ld_pair(const __bf16* p0, const __bf16* p1) {
    v8bf lo = *(const v8bf*)p0;
    v8bf hi = *(const v8bf*)p1;
    return __builtin_shufflevector(lo, hi, 0,1,2,3,4,5,6,7,8,9,10,11,12,13,14,15);
}

__device__ __forceinline__ v8f wmma_bf16(v16bf a, v16bf b, v8f c) {
    return __builtin_amdgcn_wmma_f32_16x16x32_bf16(false, a, false, b, (short)0, c,
                                                   false, false);
}

// ---------------------------------------------------------------------------
// Double-buffered 32(M) x 64(N) x KP WMMA core for one wave.
//   a0p/a1p : A rows (this lane) for the two 16-row blocks, row-major stride KP
//   bp      : B row (this lane) for N-tile 0; tiles t=0..3 at bp + t*16*KP
//   c0/kb   : lane K-offsets per the CDNA5 16-bit A/B register layouts
//   acc[8]  : acc[t] = block0 x tile t, acc[4+t] = block1 x tile t
// ---------------------------------------------------------------------------
template<int KP>
__device__ __forceinline__ void wmma_core(const __bf16* __restrict__ a0p,
                                          const __bf16* __restrict__ a1p,
                                          const __bf16* __restrict__ bp,
                                          int c0, int kb, v8f acc[8]) {
    constexpr int STEPS = KP / 32;
    const size_t TS = (size_t)16 * KP;   // B tile stride (rows of 16)

    v16bf a0 = ld_pair(a0p + c0,          a0p + c0 + 16);
    v16bf a1 = ld_pair(a1p + c0,          a1p + c0 + 16);
    v16bf b0 = ld_pair(bp + 0*TS + kb,    bp + 0*TS + kb + 16);
    v16bf b1 = ld_pair(bp + 1*TS + kb,    bp + 1*TS + kb + 16);
    v16bf b2 = ld_pair(bp + 2*TS + kb,    bp + 2*TS + kb + 16);
    v16bf b3 = ld_pair(bp + 3*TS + kb,    bp + 3*TS + kb + 16);

    #pragma unroll 4
    for (int s = 0; s < STEPS - 1; ++s) {
        const int k0 = (s + 1) * 32;
        // prefetch next k-step into separate registers
        v16bf a0n = ld_pair(a0p + k0 + c0,       a0p + k0 + c0 + 16);
        v16bf a1n = ld_pair(a1p + k0 + c0,       a1p + k0 + c0 + 16);
        v16bf b0n = ld_pair(bp + 0*TS + k0 + kb, bp + 0*TS + k0 + kb + 16);
        v16bf b1n = ld_pair(bp + 1*TS + k0 + kb, bp + 1*TS + k0 + kb + 16);
        v16bf b2n = ld_pair(bp + 2*TS + k0 + kb, bp + 2*TS + k0 + kb + 16);
        v16bf b3n = ld_pair(bp + 3*TS + k0 + kb, bp + 3*TS + k0 + kb + 16);
        // 8 WMMAs on current buffers (each B tile used twice)
        acc[0] = wmma_bf16(a0, b0, acc[0]);  acc[4] = wmma_bf16(a1, b0, acc[4]);
        acc[1] = wmma_bf16(a0, b1, acc[1]);  acc[5] = wmma_bf16(a1, b1, acc[5]);
        acc[2] = wmma_bf16(a0, b2, acc[2]);  acc[6] = wmma_bf16(a1, b2, acc[6]);
        acc[3] = wmma_bf16(a0, b3, acc[3]);  acc[7] = wmma_bf16(a1, b3, acc[7]);
        a0 = a0n; a1 = a1n; b0 = b0n; b1 = b1n; b2 = b2n; b3 = b3n;
    }
    acc[0] = wmma_bf16(a0, b0, acc[0]);  acc[4] = wmma_bf16(a1, b0, acc[4]);
    acc[1] = wmma_bf16(a0, b1, acc[1]);  acc[5] = wmma_bf16(a1, b1, acc[5]);
    acc[2] = wmma_bf16(a0, b2, acc[2]);  acc[6] = wmma_bf16(a1, b2, acc[6]);
    acc[3] = wmma_bf16(a0, b3, acc[3]);  acc[7] = wmma_bf16(a1, b3, acc[7]);
}

// ---------------------------------------------------------------- weights->bf16
__global__ __launch_bounds__(256) void conv_w_kernel(const float* __restrict__ w0,
                                                     const float* __restrict__ w1,
                                                     __bf16* __restrict__ w0b,
                                                     __bf16* __restrict__ w1b) {
    int i = blockIdx.x * 256 + threadIdx.x;
    if (i < N0 * KP0) {
        int n = i / KP0, k = i % KP0;
        w0b[i] = (__bf16)((k < FD) ? w0[n * FD + k] : 0.0f);
    }
    if (i < N1 * N0) {
        w1b[i] = (__bf16)w1[i];
    }
}

// ------------------------------------------- gather + controller + select -> z
// One wave (32 lanes) per batch row; 8 rows per 256-thread block.
__global__ __launch_bounds__(256) void controller_kernel(
    const int*   __restrict__ x,   const float* __restrict__ emb_table,
    const float* __restrict__ w_c, const float* __restrict__ b_c,
    const float* __restrict__ g_c, const float* __restrict__ be_c,
    __bf16* __restrict__ z)
{
    __shared__ float smem[8 * 704];
    const int wave = threadIdx.x >> 5;
    const int lane = threadIdx.x & 31;
    float* emb_s = smem + wave * 704;   // [39][16]  emb_s[f*16+d]
    float* sc    = emb_s + 624;         // 40 slots: scores
    float* wsel  = sc + 40;             // 40 slots: exp / selected weights
    const int b  = blockIdx.x * 8 + wave;

    // gather embeddings for this row
    for (int f = lane; f < Ff; f += 32) {
        int idx = x[b * Ff + f] + f * Vv;
        const float4* src = (const float4*)(emb_table + (size_t)idx * Dd);
        float4* dst = (float4*)(emb_s + f * Dd);
        dst[0] = src[0]; dst[1] = src[1]; dst[2] = src[2]; dst[3] = src[3];
    }
    __syncthreads();

    const float inv = rsqrtf(1.0f + 1e-5f);
    // controller dots over k = d*39 + f  (matches torch transpose(1,2) flatten)
    for (int f0 = lane; f0 < Ff; f0 += 32) {
        float acc = 0.0f;
        const float* wr = w_c + f0 * FD;
        for (int d = 0; d < Dd; ++d) {
            const float* wv = wr + d * Ff;
            #pragma unroll
            for (int f = 0; f < Ff; ++f)
                acc += emb_s[f * Dd + d] * wv[f];
        }
        float h = g_c[f0] * (acc + b_c[f0]) * inv + be_c[f0];
        sc[f0] = h > 0.0f ? h : 0.0f;
    }
    __syncthreads();

    // softmax over 39 fields
    float mx = -1e30f;
    for (int f = 0; f < Ff; ++f) mx = fmaxf(mx, sc[f]);
    for (int f0 = lane; f0 < Ff; f0 += 32) wsel[f0] = __expf(sc[f0] - mx);
    __syncthreads();
    float sum = 0.0f;
    for (int f = 0; f < Ff; ++f) sum += wsel[f];
    float rs = 1.0f / sum;
    __syncthreads();
    for (int f0 = lane; f0 < Ff; f0 += 32) sc[f0] = wsel[f0] * rs;
    __syncthreads();

    // top-K selection via rank (tie-break by lower index, like lax.top_k)
    for (int f0 = lane; f0 < Ff; f0 += 32) {
        float s = sc[f0];
        int rank = 0;
        for (int j = 0; j < Ff; ++j) {
            float sj = sc[j];
            rank += (sj > s) || (sj == s && j < f0);
        }
        wsel[f0] = (rank < Kk) ? s : 0.0f;
    }
    __syncthreads();

    // z[b, d*39+f] = emb*weight (bf16), zero-pad k in [624,640)
    __bf16* zr = z + (size_t)b * KP0;
    for (int d = 0; d < Dd; ++d)
        for (int f = lane; f < Ff; f += 32)
            zr[d * Ff + f] = (__bf16)(emb_s[f * Dd + d] * wsel[f]);
    if (lane < KP0 - FD) zr[FD + lane] = (__bf16)0.0f;
}

// -------------------------------------------------- GEMM1: z @ w0^T, BN+ReLU
// wave -> 32(M) x 64(N), K = 640. grid = (N0/64, Bsz/256), block = 256.
__global__ __launch_bounds__(256) void gemm1_kernel(
    const __bf16* __restrict__ z, const __bf16* __restrict__ w0b,
    const float* __restrict__ b0, const float* __restrict__ g0,
    const float* __restrict__ be0, __bf16* __restrict__ z1)
{
    const int wave = threadIdx.x >> 5, lane = threadIdx.x & 31;
    const int hi = lane >> 4, nl = lane & 15;
    const int c0 = hi * 8, kb = hi * 8;
    const int mbase = (blockIdx.y * 8 + wave) * 32;
    const int nbase = blockIdx.x * 64;

    const __bf16* a0p = z + (size_t)(mbase + nl) * KP0;
    const __bf16* a1p = z + (size_t)(mbase + 16 + nl) * KP0;
    const __bf16* bp  = w0b + (size_t)(nbase + nl) * KP0;

    v8f acc[8] = {v8f{}, v8f{}, v8f{}, v8f{}, v8f{}, v8f{}, v8f{}, v8f{}};
    wmma_core<KP0>(a0p, a1p, bp, c0, kb, acc);

    const float inv = rsqrtf(1.0f + 1e-5f);
    #pragma unroll
    for (int t = 0; t < 4; ++t) {
        int n = nbase + t * 16 + nl;
        float bb = b0[n], gg = g0[n], be = be0[n];
        #pragma unroll
        for (int bi = 0; bi < 2; ++bi) {
            #pragma unroll
            for (int r = 0; r < 8; ++r) {
                int mm = mbase + bi * 16 + hi * 8 + r;
                float v = gg * (acc[bi * 4 + t][r] + bb) * inv + be;
                z1[(size_t)mm * N0 + n] = (__bf16)(v > 0.0f ? v : 0.0f);
            }
        }
    }
}

// -------------------- GEMM2: z1 @ w1^T, BN+ReLU, fused w_out rank-1 partials
// grid = (N1/64, Bsz/256), block = 256. partials[m*NT2 + blockIdx.x].
__global__ __launch_bounds__(256) void gemm2_kernel(
    const __bf16* __restrict__ z1, const __bf16* __restrict__ w1b,
    const float* __restrict__ b1, const float* __restrict__ g1,
    const float* __restrict__ be1, const float* __restrict__ w_out,
    float* __restrict__ partials)
{
    const int wave = threadIdx.x >> 5, lane = threadIdx.x & 31;
    const int hi = lane >> 4, nl = lane & 15;
    const int c0 = hi * 8, kb = hi * 8;
    const int mbase = (blockIdx.y * 8 + wave) * 32;
    const int nbase = blockIdx.x * 64;

    const __bf16* a0p = z1 + (size_t)(mbase + nl) * N0;
    const __bf16* a1p = z1 + (size_t)(mbase + 16 + nl) * N0;
    const __bf16* bp  = w1b + (size_t)(nbase + nl) * N0;

    v8f acc[8] = {v8f{}, v8f{}, v8f{}, v8f{}, v8f{}, v8f{}, v8f{}, v8f{}};
    wmma_core<N0>(a0p, a1p, bp, c0, kb, acc);

    const float inv = rsqrtf(1.0f + 1e-5f);
    float p[2][8] = {{0,0,0,0,0,0,0,0},{0,0,0,0,0,0,0,0}};
    #pragma unroll
    for (int t = 0; t < 4; ++t) {
        int n = nbase + t * 16 + nl;
        float bb = b1[n], gg = g1[n], be = be1[n], wo = w_out[n];
        #pragma unroll
        for (int bi = 0; bi < 2; ++bi) {
            #pragma unroll
            for (int r = 0; r < 8; ++r) {
                float v = gg * (acc[bi * 4 + t][r] + bb) * inv + be;
                p[bi][r] += (v > 0.0f ? v : 0.0f) * wo;
            }
        }
    }
    // reduce over the 16 lanes holding the same rows (xor stays within half-wave)
    #pragma unroll
    for (int bi = 0; bi < 2; ++bi) {
        #pragma unroll
        for (int r = 0; r < 8; ++r) {
            float v = p[bi][r];
            v += __shfl_xor(v, 8, 32);
            v += __shfl_xor(v, 4, 32);
            v += __shfl_xor(v, 2, 32);
            v += __shfl_xor(v, 1, 32);
            if (nl == 0)
                partials[(size_t)(mbase + bi * 16 + hi * 8 + r) * NT2 + blockIdx.x] = v;
        }
    }
}

// ---------------------------------------------------------------- final sigmoid
__global__ __launch_bounds__(256) void final_kernel(const float* __restrict__ partials,
                                                    const float* __restrict__ b_out,
                                                    float* __restrict__ out) {
    int i = blockIdx.x * 256 + threadIdx.x;
    if (i < Bsz) {
        float s = b_out[0];
        #pragma unroll
        for (int j = 0; j < NT2; ++j) s += partials[(size_t)i * NT2 + j];
        out[i] = 1.0f / (1.0f + __expf(-s));
    }
}

extern "C" void kernel_launch(void* const* d_in, const int* in_sizes, int n_in,
                              void* d_out, int out_size, void* d_ws, size_t ws_size,
                              hipStream_t stream) {
    const int*   x         = (const int*)  d_in[0];
    const float* emb_table = (const float*)d_in[1];
    const float* w_c  = (const float*)d_in[2];
    const float* b_c  = (const float*)d_in[3];
    const float* g_c  = (const float*)d_in[4];
    const float* be_c = (const float*)d_in[5];
    const float* w0   = (const float*)d_in[6];
    const float* b0   = (const float*)d_in[7];
    const float* g0   = (const float*)d_in[8];
    const float* be0  = (const float*)d_in[9];
    const float* w1   = (const float*)d_in[10];
    const float* b1   = (const float*)d_in[11];
    const float* g1   = (const float*)d_in[12];
    const float* be1  = (const float*)d_in[13];
    const float* wout = (const float*)d_in[14];
    const float* bout = (const float*)d_in[15];
    float* out = (float*)d_out;

    // workspace layout
    __bf16* z    = (__bf16*)d_ws;                    // Bsz*KP0
    __bf16* z1   = z   + (size_t)Bsz * KP0;          // Bsz*N0
    __bf16* w0b  = z1  + (size_t)Bsz * N0;           // N0*KP0
    __bf16* w1b  = w0b + (size_t)N0 * KP0;           // N1*N0
    float*  parts = (float*)(w1b + (size_t)N1 * N0); // Bsz*NT2

    conv_w_kernel<<<(N0 * KP0 + 255) / 256, 256, 0, stream>>>(w0, w1, w0b, w1b);
    controller_kernel<<<Bsz / 8, 256, 0, stream>>>(x, emb_table, w_c, b_c, g_c, be_c, z);
    gemm1_kernel<<<dim3(N0 / 64, Bsz / 256), 256, 0, stream>>>(z, w0b, b0, g0, be0, z1);
    gemm2_kernel<<<dim3(N1 / 64, Bsz / 256), 256, 0, stream>>>(z1, w1b, b1, g1, be1, wout, parts);
    final_kernel<<<(Bsz + 255) / 256, 256, 0, stream>>>(parts, bout, out);
}
